// TransformerDecoderBlock_84198538871087
// MI455X (gfx1250) — matrix operand order repaired and checked by reading, not compiled, not run on hardware
//
#include <hip/hip_runtime.h>
#include <hip/hip_bf16.h>

// ---------------------------------------------------------------------------
// Types for CDNA5 WMMA (gfx1250, wave32): D(f32 16x16) = A(bf16 16x32) x B(bf16 32x16) + C
// ---------------------------------------------------------------------------
typedef __bf16 bf16_t;
typedef __attribute__((ext_vector_type(16))) bf16_t bf16x16;
typedef __attribute__((ext_vector_type(8)))  float  f32x8;

union FragB16 {
  bf16x16       v;
  unsigned short u[16];
  uint4          q[2];
};

__device__ __forceinline__ unsigned short f2bf(float f) {
  unsigned int u = __float_as_uint(f);
  u += 0x7fffu + ((u >> 16) & 1u);            // round-to-nearest-even
  return (unsigned short)(u >> 16);
}
__device__ __forceinline__ float bf2f(unsigned short h) {
  return __uint_as_float(((unsigned int)h) << 16);
}
__device__ __forceinline__ f32x8 zero8() {
  f32x8 z = {0.f, 0.f, 0.f, 0.f, 0.f, 0.f, 0.f, 0.f};
  return z;
}

// gfx1250 async global->LDS copy, 16B per lane, tracked by ASYNCcnt.
__device__ __forceinline__ void async_cp16(unsigned lds_addr, const void* gptr) {
  asm volatile("global_load_async_to_lds_b128 %0, %1, off"
               :: "v"(lds_addr), "v"((unsigned long long)(uintptr_t)gptr)
               : "memory");
}
__device__ __forceinline__ void wait_async_le4() {
  asm volatile("s_wait_asynccnt 0x4" ::: "memory");
}
__device__ __forceinline__ void wait_async_0() {
  asm volatile("s_wait_asynccnt 0x0" ::: "memory");
}

// ---------------------------------------------------------------------------
// Transpose + f32->bf16 conversion for weights: in[R][C] f32 -> out[C][R] bf16.
// 32x32 LDS tile, block = 256 threads (32x8).
// ---------------------------------------------------------------------------
__global__ __launch_bounds__(256)
void k_transpose_f32_bf16(const float* __restrict__ in,
                          unsigned short* __restrict__ out, int R, int C) {
  __shared__ float tile[32][33];
  const int tx = threadIdx.x & 31, ty = threadIdx.x >> 5;
  const int r0 = blockIdx.y * 32, c0 = blockIdx.x * 32;
#pragma unroll
  for (int i = 0; i < 4; ++i)
    tile[ty + i * 8][tx] = in[(size_t)(r0 + ty + i * 8) * C + c0 + tx];
  __syncthreads();
#pragma unroll
  for (int i = 0; i < 4; ++i)
    out[(size_t)(c0 + ty + i * 8) * R + r0 + tx] = f2bf(tile[tx][ty + i * 8]);
}

// ---------------------------------------------------------------------------
// RMSNorm fused with bf16 conversion. One block (256 threads) per row.
// ---------------------------------------------------------------------------
template <int D>
__global__ __launch_bounds__(256)
void k_rmsnorm_bf16(const float* __restrict__ x, const float* __restrict__ w,
                    unsigned short* __restrict__ out) {
  const int row = blockIdx.x;
  const int t   = threadIdx.x;
  const float* xr = x + (size_t)row * D;
  float s = 0.f;
#pragma unroll
  for (int i = 0; i < D / 256; ++i) {
    float v = xr[t + i * 256];
    s += v * v;
  }
#pragma unroll
  for (int off = 1; off < 32; off <<= 1) s += __shfl_xor(s, off, 32);
  __shared__ float red[8];
  if ((t & 31) == 0) red[t >> 5] = s;
  __syncthreads();
  if (t == 0) {
    float tt = 0.f;
#pragma unroll
    for (int k = 0; k < 8; ++k) tt += red[k];
    red[0] = tt;
  }
  __syncthreads();
  const float scale = rsqrtf(red[0] / (float)D + 1e-5f);
  unsigned short* orow = out + (size_t)row * D;
#pragma unroll
  for (int i = 0; i < D / 256; ++i) {
    int idx = t + i * 256;
    orow[idx] = f2bf(xr[idx] * scale * w[idx]);
  }
}

// ---------------------------------------------------------------------------
// bf16 WMMA GEMM: C[M,N] = A[M,K] @ Bt[N,K]^T with fused epilogues.
//   EPI 0: outH = bf16(acc + bias)
//   EPI 1: outF = acc + bias + residual                (f32)
//   EPI 2: outH = bf16(silu(acc + bias))
//   EPI 3: outH = bf16((acc + bias) * bf2f(other))     (GLU combine)
// B is pre-transposed [N][K] so both operands stage as contiguous 16B chunks.
// Block: 128x128 C tile, BK=32, 256 threads = 8 waves (4M x 2N), each wave a
// 32x64 C tile (2x4 WMMA accumulators). Double-buffered LDS filled with
// gfx1250 async global->LDS b128 copies (ASYNCcnt), overlapped with WMMA.
// ---------------------------------------------------------------------------
template <int EPI>
__global__ __launch_bounds__(256)
void k_gemm_bf16(const unsigned short* __restrict__ A,
                 const unsigned short* __restrict__ Bt,
                 const float* __restrict__ bias,
                 const float* __restrict__ residual,
                 const unsigned short* __restrict__ other,
                 float* __restrict__ outF,
                 unsigned short* __restrict__ outH,
                 int M, int N, int K) {
  constexpr int BM = 128, BN = 128, BK = 32;
  constexpr unsigned TILE_B = BM * BK * 2;  // 8 KB per buffer
  __shared__ __align__(16) unsigned short lA[2][BM * BK];  // M x K, row-major
  __shared__ __align__(16) unsigned short lB[2][BN * BK];  // N x K, row-major

  const int tid  = threadIdx.x;
  const int lane = tid & 31, wid = tid >> 5;
  const int wm = wid & 3, wn = wid >> 2;  // 4 x 2 wave grid
  const int bm = blockIdx.y * BM, bn = blockIdx.x * BN;

  f32x8 acc[2][4];
#pragma unroll
  for (int i = 0; i < 2; ++i)
#pragma unroll
    for (int j = 0; j < 4; ++j) acc[i][j] = zero8();

  // async staging: thread -> two 16B chunks per matrix per stage
  const int cr = tid >> 2;             // row 0..63 (and +64)
  const int cc = (tid & 3) * 8;        // col chunk: 0,8,16,24
  const unsigned lA0 = (unsigned)(uintptr_t)&lA[0][0];
  const unsigned lB0 = (unsigned)(uintptr_t)&lB[0][0];
  const unsigned stOff = (unsigned)((cr * BK + cc) * 2);
  const size_t gaRow0 = (size_t)(bm + cr) * K + cc;
  const size_t gaRow1 = (size_t)(bm + 64 + cr) * K + cc;
  const size_t gbRow0 = (size_t)(bn + cr) * K + cc;
  const size_t gbRow1 = (size_t)(bn + 64 + cr) * K + cc;

  const int khA = (lane >> 4) * 16;   // A/B fragment K runs
  const int kB  = (lane >> 4) * 16;
  const int ln  = lane & 15;

  const int nstage = K / BK;

  // prologue: stage 0 -> buffer 0
  {
    const unsigned la = lA0 + stOff, lb = lB0 + stOff;
    async_cp16(la, A + gaRow0);
    async_cp16(la + 64u * BK * 2u, A + gaRow1);
    async_cp16(lb, Bt + gbRow0);
    async_cp16(lb + 64u * BK * 2u, Bt + gbRow1);
  }

  int p = 0;
  for (int s = 0; s < nstage; ++s) {
    if (s + 1 < nstage) {  // prefetch next stage into the other buffer
      const int k0 = (s + 1) * BK;
      const unsigned la = lA0 + (unsigned)(p ^ 1) * TILE_B + stOff;
      const unsigned lb = lB0 + (unsigned)(p ^ 1) * TILE_B + stOff;
      async_cp16(la, A + gaRow0 + k0);
      async_cp16(la + 64u * BK * 2u, A + gaRow1 + k0);
      async_cp16(lb, Bt + gbRow0 + k0);
      async_cp16(lb + 64u * BK * 2u, Bt + gbRow1 + k0);
      wait_async_le4();  // stage s's 4 ops complete (per wave)
    } else {
      wait_async_0();
    }
    __syncthreads();  // stage s visible from all waves

    FragB16 af[2], bfv[4];
#pragma unroll
    for (int i = 0; i < 2; ++i) {
      const int m = wm * 32 + i * 16 + ln;
      af[i].q[0] = *(const uint4*)&lA[p][m * BK + khA];
      af[i].q[1] = *(const uint4*)&lA[p][m * BK + khA + 8];
    }
#pragma unroll
    for (int j = 0; j < 4; ++j) {
      const int n = wn * 64 + j * 16 + ln;
      bfv[j].q[0] = *(const uint4*)&lB[p][n * BK + kB];
      bfv[j].q[1] = *(const uint4*)&lB[p][n * BK + kB + 8];
    }
#pragma unroll
    for (int i = 0; i < 2; ++i)
#pragma unroll
      for (int j = 0; j < 4; ++j)
        acc[i][j] = __builtin_amdgcn_wmma_f32_16x16x32_bf16(
            false, af[i].v, false, bfv[j].v, (short)0, acc[i][j], false, false);

    __syncthreads();  // done reading buffer p before iter s+1 overwrites it
    p ^= 1;
  }

  const int rb = (lane >> 4) * 8;  // C layout: row = rb + r, col = ln
#pragma unroll
  for (int i = 0; i < 2; ++i) {
#pragma unroll
    for (int j = 0; j < 4; ++j) {
      const int col = bn + wn * 64 + j * 16 + ln;
      const float bcol_v = bias[col];
#pragma unroll
      for (int r = 0; r < 8; ++r) {
        const int row = bm + wm * 32 + i * 16 + rb + r;
        const size_t idx = (size_t)row * N + col;
        float v = acc[i][j][r] + bcol_v;
        if constexpr (EPI == 0) {
          outH[idx] = f2bf(v);
        } else if constexpr (EPI == 1) {
          outF[idx] = v + residual[idx];
        } else if constexpr (EPI == 2) {
          outH[idx] = f2bf(v / (1.f + __expf(-v)));
        } else {
          outH[idx] = f2bf(v * bf2f(other[idx]));
        }
      }
    }
  }
}

// ---------------------------------------------------------------------------
// Causal flash attention. bf16 Q/K/V stored [B,S,H*DH]; 4 waves/block,
// one wave per 16 query rows; 32-key tiles; head_dim 128 (4 WMMA K-steps).
// ---------------------------------------------------------------------------
__global__ __launch_bounds__(128)
void k_flash_attn(const unsigned short* __restrict__ Qb,
                  const unsigned short* __restrict__ Kb,
                  const unsigned short* __restrict__ Vb,
                  unsigned short* __restrict__ Ob) {
  constexpr int S = 2048, D = 2048, H = 16, DH = 128;
  const int lane = threadIdx.x & 31, wid = threadIdx.x >> 5;
  const int bh = blockIdx.y, b = bh / H, h = bh % H;
  const int q0 = blockIdx.x * 64 + wid * 16;
  const size_t headOff = (size_t)b * S * D + (size_t)h * DH;

  const int khA = (lane >> 4) * 16;
  const int kB  = (lane >> 4) * 16;
  const int ln  = lane & 15;
  const int rb  = (lane >> 4) * 8;

  // Persistent Q fragments (A-layout): lane holds row q0+ln
  FragB16 qf[4];
  {
    const size_t rowOff = headOff + (size_t)(q0 + ln) * D;
#pragma unroll
    for (int dk = 0; dk < 4; ++dk) {
      const unsigned short* p = Qb + rowOff + dk * 32 + khA;
      qf[dk].q[0] = *(const uint4*)p;
      qf[dk].q[1] = *(const uint4*)(p + 8);
    }
  }

  f32x8 o[8];
#pragma unroll
  for (int dj = 0; dj < 8; ++dj) o[dj] = zero8();
  float mrun[8], lrun[8];
#pragma unroll
  for (int r = 0; r < 8; ++r) { mrun[r] = -1e30f; lrun[r] = 0.f; }

  __shared__ __align__(16) unsigned short lp[4][16 * 32];  // per-wave P tile

  const float scale = 0.08838834764831845f;  // 1/sqrt(128)
  const int nkt = (q0 + 15) / 32 + 1;        // causal: uniform per wave

  for (int kt = 0; kt < nkt; ++kt) {
    const int kb0 = kt * 32;
    f32x8 sacc[2] = {zero8(), zero8()};
#pragma unroll
    for (int dk = 0; dk < 4; ++dk) {
#pragma unroll
      for (int j = 0; j < 2; ++j) {  // B-layout: lane holds key col kb0+j*16+ln
        FragB16 kf;
        const unsigned short* p =
            Kb + headOff + (size_t)(kb0 + j * 16 + ln) * D + dk * 32 + kB;
        kf.q[0] = *(const uint4*)p;
        kf.q[1] = *(const uint4*)(p + 8);
        sacc[j] = __builtin_amdgcn_wmma_f32_16x16x32_bf16(
            false, qf[dk].v, false, kf.v, (short)0, sacc[j], false, false);
      }
    }

    // Online softmax: rows live across 16-lane halves (C layout col = ln)
#pragma unroll
    for (int r = 0; r < 8; ++r) {
      const int qrow = q0 + rb + r;
      float s0 = sacc[0][r] * scale;
      float s1 = sacc[1][r] * scale;
      if (kb0 + ln > qrow)      s0 = -1e30f;
      if (kb0 + 16 + ln > qrow) s1 = -1e30f;
      float mx = fmaxf(s0, s1);
#pragma unroll
      for (int off = 1; off < 16; off <<= 1) mx = fmaxf(mx, __shfl_xor(mx, off, 32));
      const float mN    = fmaxf(mrun[r], mx);
      const float alpha = __expf(mrun[r] - mN);
      mrun[r] = mN;
      const float p0 = __expf(s0 - mN);
      const float p1 = __expf(s1 - mN);
      float ps = p0 + p1;
#pragma unroll
      for (int off = 1; off < 16; off <<= 1) ps += __shfl_xor(ps, off, 32);
      lrun[r] = lrun[r] * alpha + ps;
#pragma unroll
      for (int dj = 0; dj < 8; ++dj) o[dj][r] *= alpha;
      const int m = rb + r;
      lp[wid][m * 32 + ln]      = f2bf(p0);
      lp[wid][m * 32 + 16 + ln] = f2bf(p1);
    }
    asm volatile("s_wait_dscnt 0x0" ::: "memory");  // LDS relayout: C-frag -> A-frag

    FragB16 pf;
    pf.q[0] = *(const uint4*)&lp[wid][ln * 32 + khA];
    pf.q[1] = *(const uint4*)&lp[wid][ln * 32 + khA + 8];

#pragma unroll
    for (int dj = 0; dj < 8; ++dj) {  // V B-layout: K run of 16 over key rows
      FragB16 vf;
#pragma unroll
      for (int e = 0; e < 16; ++e) {
        const int t = kb0 + kB + e;
        vf.u[e] = Vb[headOff + (size_t)t * D + dj * 16 + ln];
      }
      o[dj] = __builtin_amdgcn_wmma_f32_16x16x32_bf16(
          false, pf.v, false, vf.v, (short)0, o[dj], false, false);
    }
  }

#pragma unroll
  for (int dj = 0; dj < 8; ++dj) {
#pragma unroll
    for (int r = 0; r < 8; ++r) {
      const int qrow = q0 + rb + r;
      Ob[headOff + (size_t)qrow * D + dj * 16 + ln] = f2bf(o[dj][r] / lrun[r]);
    }
  }
}

// ---------------------------------------------------------------------------
// Launch sequence
// ---------------------------------------------------------------------------
extern "C" void kernel_launch(void* const* d_in, const int* in_sizes, int n_in,
                              void* d_out, int out_size, void* d_ws, size_t ws_size,
                              hipStream_t stream) {
  constexpr int Dm = 2048, P = 8192, S = 2048, Bb = 2, HEADS = 16;
  constexpr int Mr = Bb * S;  // 4096 rows

  const float* x   = (const float*)d_in[0];
  const float* ln1 = (const float*)d_in[1];
  const float* ln2 = (const float*)d_in[2];
  const float* wq  = (const float*)d_in[3];
  const float* bq  = (const float*)d_in[4];
  const float* wk  = (const float*)d_in[5];
  const float* bk  = (const float*)d_in[6];
  const float* wv  = (const float*)d_in[7];
  const float* bv  = (const float*)d_in[8];
  const float* wo  = (const float*)d_in[9];
  const float* bo  = (const float*)d_in[10];
  const float* w1  = (const float*)d_in[11];
  const float* b1  = (const float*)d_in[12];
  const float* w2  = (const float*)d_in[13];
  const float* b2  = (const float*)d_in[14];
  const float* w3  = (const float*)d_in[15];
  const float* b3  = (const float*)d_in[16];

  char* ws = (char*)d_ws;
  size_t off = 0;
  auto alloc = [&](size_t bytes) -> char* {
    char* p = ws + off;
    off += (bytes + 255) & ~(size_t)255;
    return p;
  };
  // all weights stored TRANSPOSED [N][K] in bf16
  unsigned short* Wq = (unsigned short*)alloc((size_t)Dm * Dm * 2);
  unsigned short* Wk = (unsigned short*)alloc((size_t)Dm * Dm * 2);
  unsigned short* Wv = (unsigned short*)alloc((size_t)Dm * Dm * 2);
  unsigned short* Wo = (unsigned short*)alloc((size_t)Dm * Dm * 2);
  unsigned short* W1 = (unsigned short*)alloc((size_t)Dm * P * 2);
  unsigned short* W2 = (unsigned short*)alloc((size_t)Dm * P * 2);
  unsigned short* W3 = (unsigned short*)alloc((size_t)P * Dm * 2);
  unsigned short* Xn = (unsigned short*)alloc((size_t)Mr * Dm * 2);
  unsigned short* Qb = (unsigned short*)alloc((size_t)Mr * Dm * 2);
  unsigned short* Kb = (unsigned short*)alloc((size_t)Mr * Dm * 2);
  unsigned short* Vb = (unsigned short*)alloc((size_t)Mr * Dm * 2);
  unsigned short* Ob = (unsigned short*)alloc((size_t)Mr * Dm * 2);
  float*          Y1 = (float*)alloc((size_t)Mr * Dm * 4);
  unsigned short* Hb = (unsigned short*)alloc((size_t)Mr * P * 2);
  unsigned short* A1 = Qb;  // alias: Q/K/V/O span == Mr*P*2 bytes, dead after Wo-proj

  auto tr = [&](const float* src, unsigned short* dst, int R, int C) {
    // src: R x C f32 -> dst: C x R bf16
    k_transpose_f32_bf16<<<dim3(C / 32, R / 32), 256, 0, stream>>>(src, dst, R, C);
  };
  tr(wq, Wq, Dm, Dm);   // [Dm][Dm] -> Wq[N=Dm][K=Dm]
  tr(wk, Wk, Dm, Dm);
  tr(wv, Wv, Dm, Dm);
  tr(wo, Wo, Dm, Dm);
  tr(w1, W1, Dm, P);    // [Dm][P] -> W1[N=P][K=Dm]
  tr(w2, W2, Dm, P);
  tr(w3, W3, P, Dm);    // [P][Dm] -> W3[N=Dm][K=P]

  const dim3 blk(256);
  const dim3 gDD(Dm / 128, Mr / 128);
  const dim3 gDP(P / 128, Mr / 128);

  // x1n = rmsnorm(x, ln1)
  k_rmsnorm_bf16<Dm><<<Mr, blk, 0, stream>>>(x, ln1, Xn);
  // Q, K, V projections
  k_gemm_bf16<0><<<gDD, blk, 0, stream>>>(Xn, Wq, bq, nullptr, nullptr, nullptr, Qb, Mr, Dm, Dm);
  k_gemm_bf16<0><<<gDD, blk, 0, stream>>>(Xn, Wk, bk, nullptr, nullptr, nullptr, Kb, Mr, Dm, Dm);
  k_gemm_bf16<0><<<gDD, blk, 0, stream>>>(Xn, Wv, bv, nullptr, nullptr, nullptr, Vb, Mr, Dm, Dm);
  // causal flash attention
  k_flash_attn<<<dim3(S / 64, Bb * HEADS), dim3(128), 0, stream>>>(Qb, Kb, Vb, Ob);
  // y1 = x + O @ wo + bo
  k_gemm_bf16<1><<<gDD, blk, 0, stream>>>(Ob, Wo, bo, x, nullptr, Y1, nullptr, Mr, Dm, Dm);
  // x2n = rmsnorm(y1, ln2)
  k_rmsnorm_bf16<Dm><<<Mr, blk, 0, stream>>>(Y1, ln2, Xn);
  // A1 = silu(x2n @ w1 + b1)
  k_gemm_bf16<2><<<gDP, blk, 0, stream>>>(Xn, W1, b1, nullptr, nullptr, nullptr, A1, Mr, P, Dm);
  // H = A1 * (x2n @ w2 + b2)
  k_gemm_bf16<3><<<gDP, blk, 0, stream>>>(Xn, W2, b2, nullptr, A1, nullptr, Hb, Mr, P, Dm);
  // out = y1 + H @ w3 + b3
  k_gemm_bf16<1><<<gDD, blk, 0, stream>>>(Hb, W3, b3, Y1, nullptr, (float*)d_out, nullptr, Mr, Dm, P);
}
